// Attention_11647951307404
// MI455X (gfx1250) — compile-verified, matrix operand order checked by reading, not run on hardware
//
#include <hip/hip_runtime.h>

typedef __attribute__((ext_vector_type(16))) _Float16 v16h;
typedef __attribute__((ext_vector_type(8)))  _Float16 v8h;
typedef __attribute__((ext_vector_type(4)))  _Float16 v4h;
typedef __attribute__((ext_vector_type(8)))  float    v8f;
typedef __attribute__((ext_vector_type(4)))  float    v4f;
typedef __attribute__((ext_vector_type(4)))  unsigned int u32x4;
typedef __attribute__((ext_vector_type(8)))  unsigned int u32x8;

#define WMMA_F32_F16(a, b, c) \
    __builtin_amdgcn_wmma_f32_16x16x32_f16(false, (a), false, (b), (short)0, (c), false, false)

static __device__ __forceinline__ v16h cat16(v8h lo, v8h hi) {
    return __builtin_shufflevector(lo, hi, 0,1,2,3,4,5,6,7,8,9,10,11,12,13,14,15);
}

// ---------------------------------------------------------------------------
// TDM tile descriptor (ISA 8.3/8.4): 2D tile, 128 rows x 32 f16, row stride
// 2048 elements, data_size=2B.  Group1 is tile-geometry (loop-invariant);
// Group0 carries lds_addr + global_addr (+ type=2, count=1).
// ---------------------------------------------------------------------------
static __device__ __forceinline__ u32x8 tdm_group1_128x32(void) {
    u32x8 g1;
    g1[0] = 0x00010000u;            // wg_mask=0, data_size=1 (2 bytes)
    g1[1] = 0u;                     // abar=0; tensor_dim0[15:0]=0  (dim0=1<<20)
    g1[2] = 16u;                    // tensor_dim0[31:16]=16; tensor_dim1[15:0]=0
    g1[3] = 16u | (32u << 16);      // tensor_dim1[31:16]=16 (1<<20); tile_dim0=32
    g1[4] = 128u;                   // tile_dim1=128; tile_dim2=0 (2D)
    g1[5] = 2048u;                  // tensor_dim0_stride = 2048 elements
    g1[6] = 0u;
    g1[7] = 0u;
    return g1;
}

static __device__ __forceinline__ void tdm_load_tile(
    unsigned int lds_off, const _Float16* gptr, u32x8 g1)
{
    unsigned long long ga = (unsigned long long)(uintptr_t)gptr;
    u32x4 g0;
    g0[0] = 1u;                                            // count=1 (user D#)
    g0[1] = lds_off;                                       // LDS byte address
    g0[2] = (unsigned int)ga;                              // global_addr[31:0]
    g0[3] = (unsigned int)((ga >> 32) & 0x01FFFFFFu) | 0x80000000u; // [56:32]+type=2
    asm volatile("tensor_load_to_lds %0, %1" :: "s"(g0), "s"(g1) : "memory");
}

// ---------------------------------------------------------------------------
// GEMM: Out[M,2048] = Xh[M,2048] @ Wt[2048,2048]^T  (Wt stored [n][k] f16).
// Block = 256 threads (8 waves), tile 128x128, K-step 32, TDM double-buffered
// staging overlapped with WMMA via TENSORcnt.  N = K = 2048 compile-time.
// ---------------------------------------------------------------------------
template <typename OutT>
__global__ __launch_bounds__(256) void gemm_tdm(
    const _Float16* __restrict__ Xh, const _Float16* __restrict__ Wt,
    OutT* __restrict__ Out, int M)
{
    __shared__ __align__(16) _Float16 lds_a[2][128 * 32];
    __shared__ __align__(16) _Float16 lds_b[2][128 * 32];

    const int tid  = threadIdx.x;
    const int lane = tid & 31;
    const int wave = tid >> 5;
    const int wm   = wave >> 2;          // 0..1 (M)
    const int wn   = wave & 3;           // 0..3 (N)
    const int l    = lane & 15;
    const int hi   = lane >> 4;
    const int bm   = blockIdx.y * 128;
    const int bn   = blockIdx.x * 128;
    const int NT   = 2048 / 32;

    const unsigned int a_off[2] = { (unsigned int)(uintptr_t)&lds_a[0][0],
                                    (unsigned int)(uintptr_t)&lds_a[1][0] };
    const unsigned int b_off[2] = { (unsigned int)(uintptr_t)&lds_b[0][0],
                                    (unsigned int)(uintptr_t)&lds_b[1][0] };
    const _Float16* ag = Xh + (size_t)bm * 2048;   // tile rows, stride 2048
    const _Float16* bg = Wt + (size_t)bn * 2048;
    const u32x8 g1 = tdm_group1_128x32();

    v8f c[4][2];
#pragma unroll
    for (int i = 0; i < 4; ++i)
#pragma unroll
        for (int j = 0; j < 2; ++j) c[i][j] = (v8f)0.0f;

    if (wave == 0) {                                 // prologue: tile 0
        tdm_load_tile(a_off[0], ag, g1);
        tdm_load_tile(b_off[0], bg, g1);
    }

    for (int it = 0; it < NT; ++it) {
        const int cur = it & 1;
        if (wave == 0) {
            if (it + 1 < NT) {                       // prefetch next k-tile
                tdm_load_tile(a_off[cur ^ 1], ag + (it + 1) * 32, g1);
                tdm_load_tile(b_off[cur ^ 1], bg + (it + 1) * 32, g1);
                __builtin_amdgcn_s_wait_tensorcnt((short)2);  // cur pair done
            } else {
                __builtin_amdgcn_s_wait_tensorcnt((short)0);
            }
        }
        __syncthreads();

#pragma unroll
        for (int mf = 0; mf < 4; ++mf) {
            const _Float16* ap = &lds_a[cur][(wm * 64 + mf * 16 + l) * 32 + 8 * hi];
            v16h a = cat16(*(const v8h*)ap, *(const v8h*)(ap + 16));
#pragma unroll
            for (int nf = 0; nf < 2; ++nf) {
                const _Float16* bp = &lds_b[cur][(wn * 32 + nf * 16 + l) * 32 + 8 * hi];
                v16h b = cat16(*(const v8h*)bp, *(const v8h*)(bp + 16));
                c[mf][nf] = WMMA_F32_F16(a, b, c[mf][nf]);
            }
        }
        __syncthreads();   // protect buffer before it is overwritten next iter
    }

    // writeout: one 64-bit base, all offsets compile-time immediates
    OutT* obase = Out + (size_t)(bm + wm * 64 + 8 * hi) * 2048 + bn + wn * 32 + l;
#pragma unroll
    for (int mf = 0; mf < 4; ++mf)
#pragma unroll
        for (int nf = 0; nf < 2; ++nf)
#pragma unroll
            for (int r = 0; r < 8; ++r)
                obase[(mf * 16 + r) * 2048 + nf * 16] = (OutT)c[mf][nf][r];
}

// ---------------------------------------------------------------------------
// f32 -> f16 bulk convert (8 elems/thread, b128 in / b128 out)
// ---------------------------------------------------------------------------
__global__ __launch_bounds__(256) void pack_f16(
    const float* __restrict__ in, _Float16* __restrict__ out)
{
    size_t i = ((size_t)blockIdx.x * 256 + threadIdx.x) * 8;
    const v4f* s = (const v4f*)(in + i);
    v4f f0 = s[0], f1 = s[1];
    v8h h;
#pragma unroll
    for (int j = 0; j < 4; ++j) { h[j] = (_Float16)f0[j]; h[4 + j] = (_Float16)f1[j]; }
    *(v8h*)(out + i) = h;
}

// ---------------------------------------------------------------------------
// W[K,N] f32 -> Wt[N,K] f16, LDS-tiled 64x64 transpose (coalesced both ways)
// ---------------------------------------------------------------------------
__global__ __launch_bounds__(256) void transpose_f16t(
    const float* __restrict__ W, _Float16* __restrict__ Wt)
{
    __shared__ _Float16 t[64 * 65];
    const int tid = threadIdx.x;
    const int tk  = blockIdx.y * 64;     // K tile
    const int tn  = blockIdx.x * 64;     // N tile
    const int rr  = tid >> 4;            // 0..15
    const int cc  = (tid & 15) * 4;      // 0..60

#pragma unroll
    for (int i = 0; i < 4; ++i) {
        int r = rr + i * 16;             // k within tile
        v4f f = *(const v4f*)(W + (size_t)(tk + r) * 2048 + tn + cc);
#pragma unroll
        for (int j = 0; j < 4; ++j) t[r * 65 + cc + j] = (_Float16)f[j];
    }
    __syncthreads();
#pragma unroll
    for (int i = 0; i < 4; ++i) {
        int n = rr + i * 16;             // n within tile
        v4h o;
#pragma unroll
        for (int j = 0; j < 4; ++j) o[j] = t[(cc + j) * 65 + n];
        *(v4h*)(Wt + (size_t)(tn + n) * 2048 + tk + cc) = o;
    }
}

// ---------------------------------------------------------------------------
// RoPE + repack (f16 in / f16 out).  B=2, S=2048, heads=16, Hd=128.
//   q16,k16: [B, head, S, 128] (k pre-scaled by 128^-0.5)
//   vt16   : [B, head, 128, S]
// ---------------------------------------------------------------------------
__global__ __launch_bounds__(256) void rope_pack(
    const _Float16* __restrict__ qh, const _Float16* __restrict__ kh,
    const _Float16* __restrict__ vh, const float* __restrict__ sin_t,
    const float* __restrict__ cos_t,
    _Float16* __restrict__ q16, _Float16* __restrict__ k16,
    _Float16* __restrict__ vt16)
{
    int idx  = blockIdx.x * blockDim.x + threadIdx.x;    // over B*S*D = 2^23
    int h    = idx & 127;
    int head = (idx >> 7) & 15;
    int s    = (idx >> 11) & 2047;
    int b    = idx >> 22;

    size_t src = ((size_t)(b * 2048 + s)) * 2048 + head * 128 + h;
    float  cs  = cos_t[s * 128 + h];
    float  sn  = sin_t[s * 128 + h];
    size_t rot = (h < 64) ? src + 64 : src - 64;
    float  sgn = (h < 64) ? -1.0f : 1.0f;

    float qr = (float)qh[src] * cs + sgn * (float)qh[rot] * sn;
    float kr = ((float)kh[src] * cs + sgn * (float)kh[rot] * sn)
               * 0.08838834764831845f;                   // 128^-0.5

    size_t d1 = ((size_t)((b * 16 + head) * 2048 + s)) * 128 + h;
    q16[d1] = (_Float16)qr;
    k16[d1] = (_Float16)kr;
    vt16[((size_t)((b * 16 + head) * 128 + h)) * 2048 + s] = vh[src];
}

// ---------------------------------------------------------------------------
// Flash attention: 1 wave per 16-query tile, kv in tiles of 32, causal
// early-exit.  All fragment loads of a stage are issued as one clause into
// distinct registers so WMMAs retire them with partial LOADcnt waits.
// ---------------------------------------------------------------------------
__global__ __launch_bounds__(128) void flash_attn(
    const _Float16* __restrict__ q16, const _Float16* __restrict__ k16,
    const _Float16* __restrict__ vt16, _Float16* __restrict__ yh)
{
    __shared__ __align__(16) _Float16 pbuf[4][16 * 32];

    const int lane = threadIdx.x & 31;
    const int wave = threadIdx.x >> 5;
    const int l    = lane & 15;
    const int hi   = lane >> 4;

    const int tile = blockIdx.x * 4 + wave;
    const int st   = tile & 127;
    const int head = (tile >> 7) & 15;
    const int b    = tile >> 11;
    const int s0   = st * 16;

    const _Float16* qbase = q16  + ((size_t)((b * 16 + head) * 2048)) * 128;
    const _Float16* kbase = k16  + ((size_t)((b * 16 + head) * 2048)) * 128;
    const _Float16* vbase = vt16 + ((size_t)((b * 16 + head) * 128)) * 2048;

    v16h aq[4];
#pragma unroll
    for (int j = 0; j < 4; ++j) {
        const _Float16* p = qbase + (size_t)(s0 + l) * 128 + j * 32 + 8 * hi;
        aq[j] = cat16(*(const v8h*)p, *(const v8h*)(p + 16));
    }

    v8f acc[8];
#pragma unroll
    for (int i = 0; i < 8; ++i) acc[i] = (v8f)0.0f;
    float mrow[8], lrow[8];
#pragma unroll
    for (int r = 0; r < 8; ++r) { mrow[r] = -1e30f; lrow[r] = 0.0f; }

    _Float16* pb = pbuf[wave];
    const int ntiles = ((s0 + 15) >> 5) + 1;

    for (int it = 0; it < ntiles; ++it) {
        const int t0 = it * 32;
        if (it + 1 < ntiles) {                       // next-tile prefetch
            __builtin_prefetch(kbase + (size_t)(t0 + 32 + lane) * 128, 0, 1);
            __builtin_prefetch(vbase + (size_t)(lane * 4) * 2048 + t0 + 32, 0, 1);
        }

        // ---- stage 1: batch all 8 K fragments, then 8 score WMMAs --------
        v16h bk[2][4];
#pragma unroll
        for (int sub = 0; sub < 2; ++sub)
#pragma unroll
            for (int j = 0; j < 4; ++j) {
                const _Float16* p =
                    kbase + (size_t)(t0 + sub * 16 + l) * 128 + j * 32 + 8 * hi;
                bk[sub][j] = cat16(*(const v8h*)p, *(const v8h*)(p + 16));
            }
        v8f sc[2];
#pragma unroll
        for (int sub = 0; sub < 2; ++sub) {
            v8f s = (v8f)0.0f;
#pragma unroll
            for (int j = 0; j < 4; ++j) s = WMMA_F32_F16(aq[j], bk[sub][j], s);
            sc[sub] = s;
        }

        // ---- causal mask + streaming softmax -----------------------------
        float mnew[8];
#pragma unroll
        for (int r = 0; r < 8; ++r) {
            int m = s0 + r + 8 * hi;
            if (t0 + l      > m) sc[0][r] += -1e9f;
            if (t0 + 16 + l > m) sc[1][r] += -1e9f;
            float v = fmaxf(sc[0][r], sc[1][r]);
#pragma unroll
            for (int d = 1; d < 16; d <<= 1) v = fmaxf(v, __shfl_xor(v, d, 32));
            mnew[r] = fmaxf(mrow[r], v);
        }
#pragma unroll
        for (int r = 0; r < 8; ++r) {
            float p0 = __expf(sc[0][r] - mnew[r]);
            float p1 = __expf(sc[1][r] - mnew[r]);
            sc[0][r] = p0; sc[1][r] = p1;
            float rs = p0 + p1;
#pragma unroll
            for (int d = 1; d < 16; d <<= 1) rs += __shfl_xor(rs, d, 32);
            float scale = __expf(mrow[r] - mnew[r]);
            lrow[r] = lrow[r] * scale + rs;
            mrow[r] = mnew[r];
#pragma unroll
            for (int hc = 0; hc < 8; ++hc) acc[hc][r] *= scale;
        }

        // ---- stage 2: LDS P-transpose overlapped with batched V loads ----
#pragma unroll
        for (int sub = 0; sub < 2; ++sub)
#pragma unroll
            for (int r = 0; r < 8; ++r)
                pb[(r + 8 * hi) * 32 + sub * 16 + l] = (_Float16)sc[sub][r];

        v16h bv[8];                                  // V loads fly during repack
#pragma unroll
        for (int hc = 0; hc < 8; ++hc) {
            const _Float16* vp = vbase + (size_t)(hc * 16 + l) * 2048 + t0 + 8 * hi;
            bv[hc] = cat16(*(const v8h*)vp, *(const v8h*)(vp + 16));
        }

        asm volatile("s_wait_dscnt 0" ::: "memory");
        const _Float16* pp = pb + l * 32 + 8 * hi;
        v16h ap = cat16(*(const v8h*)pp, *(const v8h*)(pp + 16));
        asm volatile("" ::: "memory");

#pragma unroll
        for (int hc = 0; hc < 8; ++hc)
            acc[hc] = WMMA_F32_F16(ap, bv[hc], acc[hc]);
    }

    // immediate-offset writeout from a single base
    _Float16* yb = yh + ((size_t)(b * 2048 + s0 + 8 * hi)) * 2048 + head * 128 + l;
#pragma unroll
    for (int r = 0; r < 8; ++r) {
        float rinv = __builtin_amdgcn_rcpf(lrow[r]);
#pragma unroll
        for (int hc = 0; hc < 8; ++hc)
            yb[r * 2048 + hc * 16] = (_Float16)(acc[hc][r] * rinv);
    }
}

// ---------------------------------------------------------------------------
extern "C" void kernel_launch(void* const* d_in, const int* in_sizes, int n_in,
                              void* d_out, int out_size, void* d_ws, size_t ws_size,
                              hipStream_t stream)
{
    const float* x     = (const float*)d_in[0];
    const float* wq    = (const float*)d_in[1];
    const float* wk    = (const float*)d_in[2];
    const float* wv    = (const float*)d_in[3];
    const float* wo    = (const float*)d_in[4];
    // d_in[5] = mask: causal -1e9 mask computed analytically in-kernel
    const float* sin_t = (const float*)d_in[6];
    const float* cos_t = (const float*)d_in[7];
    float* out = (float*)d_out;

    const int    M = 4096;                     // B*S
    const size_t E = (size_t)M * 2048;         // elements per [B*S, D] matrix

    char*     ws   = (char*)d_ws;
    _Float16* Xh   = (_Float16*)(ws);              // x (f16); reused as yh later
    _Float16* Wt   = (_Float16*)(ws + E * 2);      // current transposed weight
    _Float16* qh   = (_Float16*)(ws + E * 4);
    _Float16* kh   = (_Float16*)(ws + E * 6);
    _Float16* vh   = (_Float16*)(ws + E * 8);
    _Float16* q16  = (_Float16*)(ws + E * 10);
    _Float16* k16  = (_Float16*)(ws + E * 12);
    _Float16* vt16 = (_Float16*)(ws + E * 14);
    _Float16* yhb  = Xh;

    dim3 gblk(2048 / 128, M / 128);            // 16 x 32
    dim3 tblk(2048 / 64, 2048 / 64);           // 32 x 32

    pack_f16<<<(int)(E / 8 / 256), 256, 0, stream>>>(x, Xh);

    transpose_f16t<<<tblk, 256, 0, stream>>>(wq, Wt);
    gemm_tdm<_Float16><<<gblk, 256, 0, stream>>>(Xh, Wt, qh, M);
    transpose_f16t<<<tblk, 256, 0, stream>>>(wk, Wt);
    gemm_tdm<_Float16><<<gblk, 256, 0, stream>>>(Xh, Wt, kh, M);
    transpose_f16t<<<tblk, 256, 0, stream>>>(wv, Wt);
    gemm_tdm<_Float16><<<gblk, 256, 0, stream>>>(Xh, Wt, vh, M);

    rope_pack<<<(int)(E / 256), 256, 0, stream>>>(qh, kh, vh, sin_t, cos_t,
                                                  q16, k16, vt16);

    flash_attn<<<2 * 16 * (2048 / 16) / 4, 128, 0, stream>>>(q16, k16, vt16, yhb);

    transpose_f16t<<<tblk, 256, 0, stream>>>(wo, Wt);
    gemm_tdm<float><<<gblk, 256, 0, stream>>>(yhb, Wt, out, M);
}
